// GEAETLayer_30992484008124
// MI455X (gfx1250) — matrix operand order repaired and checked by reading, not compiled, not run on hardware
//
#include <hip/hip_runtime.h>
#include <math.h>

#define GCNT   256
#define NPG_   192
#define DIM    128
#define ECNT   786432
#define HEADS  8
#define NNODES (GCNT * NPG_)          // 49152
#define BN_INV 0.99999500003749968f   // 1/sqrt(1 + 1e-5)

typedef __attribute__((ext_vector_type(16))) _Float16 v16h;
typedef __attribute__((ext_vector_type(8)))  _Float16 v8h;
typedef __attribute__((ext_vector_type(8)))  float    v8f;

// ---------------------------------------------------------------------------
// GEMM:  out[N,M] = A[N,K](f16) @ W[M,K](f16)^T + bias(f32), optional ReLU.
// Writes f32 and/or f16 outputs (either pointer may be null; branch uniform).
//
// Block = 128 threads (4 waves). Grid = (Nrows/64, M/64): the 4 waves cover
// 4 row-tiles of the SAME 64-wide column strip, so the weight panel
// W[colBase..colBase+64, 0..K) is staged once into LDS and shared.
// Staging uses CDNA5 GLOBAL_LOAD_ASYNC_TO_LDS_B128 (ASYNCcnt) — each lane
// moves 16B global->LDS without VGPR data movement.
//
// Fragment layouts used by this code (wave32, contiguous k order):
//  A 16x32 f16: lane holds row (l&15); half=l>>4: K run [half*16,+16)
//               -> two contiguous 16B loads.
//  B 32x16 f16: lane holds col (l&15); kh=l>>4: K run [kh*16,+16)
//               -> one contiguous 32B LDS read (2x ds_load_b128).
//  D: row = vgpr + 8*half, col = lane&15.
// ---------------------------------------------------------------------------
__global__ void gemm_wmma(const _Float16* __restrict__ A, const _Float16* __restrict__ W,
                          const float* __restrict__ bias, float* __restrict__ out32,
                          _Float16* __restrict__ out16, int K, int M, int do_relu) {
    __shared__ _Float16 sW[64 * 256];   // up to K=256

    const int lane = threadIdx.x & 31;
    const int wave = threadIdx.x >> 5;
    const int rowBase = blockIdx.x * 64 + wave * 16;
    const int colBase = blockIdx.y * 64;

    // ---- async-stage weight panel (64*K f16, contiguous in W) into LDS ----
    {
        const unsigned long long gsrc =
            (unsigned long long)(W + (size_t)colBase * K);
        const unsigned ldsbase = (unsigned)(uintptr_t)&sW[0];
        const int bytes = 64 * K * 2;
        for (int ofs = threadIdx.x * 16; ofs < bytes; ofs += 128 * 16) {
            unsigned dst  = ldsbase + (unsigned)ofs;
            unsigned voff = (unsigned)ofs;
            asm volatile("global_load_async_to_lds_b128 %0, %1, %2"
                         :: "v"(dst), "v"(voff), "s"(gsrc)
                         : "memory");
        }
        asm volatile("s_wait_asynccnt 0x0" ::: "memory");
    }
    __syncthreads();

    v8f acc0 = {}, acc1 = {}, acc2 = {}, acc3 = {};

    const int m    = lane & 15;
    const int half = lane >> 4;
    const _Float16* aptr = A + (size_t)(rowBase + m) * K + half * 16;
    const int nl = lane & 15;
    const int kh = lane >> 4;

    for (int k = 0; k < K; k += 32) {
        __builtin_prefetch(aptr + k + 32, 0, 0);   // speculative, safe
        const v8h alo = *reinterpret_cast<const v8h*>(aptr + k);
        const v8h ahi = *reinterpret_cast<const v8h*>(aptr + k + 8);
        const v16h a  = __builtin_shufflevector(alo, ahi,
                          0,1,2,3,4,5,6,7,8,9,10,11,12,13,14,15);
        const int bofs = k + kh * 16;
        const v16h b0 = *reinterpret_cast<const v16h*>(&sW[( 0 + nl) * K + bofs]);
        const v16h b1 = *reinterpret_cast<const v16h*>(&sW[(16 + nl) * K + bofs]);
        const v16h b2 = *reinterpret_cast<const v16h*>(&sW[(32 + nl) * K + bofs]);
        const v16h b3 = *reinterpret_cast<const v16h*>(&sW[(48 + nl) * K + bofs]);
        acc0 = __builtin_amdgcn_wmma_f32_16x16x32_f16(false, a, false, b0, (short)0, acc0, false, false);
        acc1 = __builtin_amdgcn_wmma_f32_16x16x32_f16(false, a, false, b1, (short)0, acc1, false, false);
        acc2 = __builtin_amdgcn_wmma_f32_16x16x32_f16(false, a, false, b2, (short)0, acc2, false, false);
        acc3 = __builtin_amdgcn_wmma_f32_16x16x32_f16(false, a, false, b3, (short)0, acc3, false, false);
    }

    const float bv0 = bias[colBase + nl];
    const float bv1 = bias[colBase + 16 + nl];
    const float bv2 = bias[colBase + 32 + nl];
    const float bv3 = bias[colBase + 48 + nl];
#pragma unroll
    for (int r = 0; r < 8; ++r) {
        const int row = rowBase + half * 8 + r;
        float v0 = acc0[r] + bv0;
        float v1 = acc1[r] + bv1;
        float v2 = acc2[r] + bv2;
        float v3 = acc3[r] + bv3;
        if (do_relu) {
            v0 = fmaxf(v0, 0.f); v1 = fmaxf(v1, 0.f);
            v2 = fmaxf(v2, 0.f); v3 = fmaxf(v3, 0.f);
        }
        if (out32) {
            float* crow = out32 + (size_t)row * M + colBase + nl;
            crow[0] = v0; crow[16] = v1; crow[32] = v2; crow[48] = v3;
        }
        if (out16) {
            _Float16* hrow = out16 + (size_t)row * M + colBase + nl;
            hrow[0]  = (_Float16)v0; hrow[16] = (_Float16)v1;
            hrow[32] = (_Float16)v2; hrow[48] = (_Float16)v3;
        }
    }
}

// ---------------------------------------------------------------------------
// Edge stage: aggr[dst] += relu(x[src] + edge_attr). 32 lanes/edge, float4.
// ---------------------------------------------------------------------------
__global__ void edge_scatter(const float* __restrict__ x, const float* __restrict__ ea,
                             const int* __restrict__ ei, float* __restrict__ aggr) {
    const long tid = (long)blockIdx.x * blockDim.x + threadIdx.x;
    const int e  = (int)(tid >> 5);
    const int c4 = (int)(tid & 31) << 2;
    const int src = ei[e];
    const int dst = ei[ECNT + e];
    const float4 xv = *reinterpret_cast<const float4*>(x  + (size_t)src * DIM + c4);
    const float4 ev = *reinterpret_cast<const float4*>(ea + (size_t)e   * DIM + c4);
    float* base = aggr + (size_t)dst * DIM + c4;
    atomicAdd(base + 0, fmaxf(xv.x + ev.x, 0.f));
    atomicAdd(base + 1, fmaxf(xv.y + ev.y, 0.f));
    atomicAdd(base + 2, fmaxf(xv.z + ev.z, 0.f));
    atomicAdd(base + 3, fmaxf(xv.w + ev.w, 0.f));
}

// ---------------------------------------------------------------------------
// Per-(graph,head) attention, online softmax, K/V staged in LDS (24 KB).
// qkv is f16: rows [q(128)|k(128)|v(128)]; head slice = h*16..+16.
// o stored f16 as [G, n, H, dh] -> o[row*128 + h*16 + d].
// ---------------------------------------------------------------------------
__global__ void attn_kernel(const _Float16* __restrict__ qkv, _Float16* __restrict__ o) {
    const int g = blockIdx.x / HEADS;
    const int h = blockIdx.x % HEADS;
    const int i = threadIdx.x;              // 0..191 (query row)
    __shared__ float sk[NPG_ * 16];
    __shared__ float sv[NPG_ * 16];
    const size_t rowBase = (size_t)g * NPG_;

    const _Float16* kp = qkv + (rowBase + i) * 384 + 128 + h * 16;
    const _Float16* vp = qkv + (rowBase + i) * 384 + 256 + h * 16;
#pragma unroll
    for (int d = 0; d < 16; ++d) { sk[i * 16 + d] = (float)kp[d]; sv[i * 16 + d] = (float)vp[d]; }

    float q[16];
    const _Float16* qp = qkv + (rowBase + i) * 384 + h * 16;
#pragma unroll
    for (int d = 0; d < 16; ++d) q[d] = (float)qp[d];
    __syncthreads();

    float mx = -1e30f, l = 0.f, acc[16];
#pragma unroll
    for (int d = 0; d < 16; ++d) acc[d] = 0.f;

    for (int j = 0; j < NPG_; ++j) {
        float s = 0.f;
#pragma unroll
        for (int d = 0; d < 16; ++d) s += q[d] * sk[j * 16 + d];
        s *= 0.25f;                                  // 1/sqrt(dh), dh=16
        const float mn   = fmaxf(mx, s);
        const float corr = __expf(mx - mn);
        const float p    = __expf(s - mn);
        l = l * corr + p;
#pragma unroll
        for (int d = 0; d < 16; ++d) acc[d] = acc[d] * corr + p * sv[j * 16 + d];
        mx = mn;
    }
    const float invl = 1.f / l;
    _Float16* op = o + (rowBase + i) * DIM + h * 16;
#pragma unroll
    for (int d = 0; d < 16; ++d) op[d] = (_Float16)(acc[d] * invl);
}

// ---------------------------------------------------------------------------
// Elementwise helpers
// ---------------------------------------------------------------------------
__global__ void zero_kernel(float* __restrict__ p, int n) {
    const int i = blockIdx.x * blockDim.x + threadIdx.x;
    if (i < n) p[i] = 0.f;
}

__global__ void cvt16(const float* __restrict__ in, _Float16* __restrict__ out, int n) {
    const int i = blockIdx.x * blockDim.x + threadIdx.x;
    if (i < n) out[i] = (_Float16)in[i];
}

// z16 = f16(a + b)
__global__ void add2_f16(const float* __restrict__ a, const float* __restrict__ b,
                         _Float16* __restrict__ c, int n) {
    const int i = blockIdx.x * blockDim.x + threadIdx.x;
    if (i < n) c[i] = (_Float16)(a[i] + b[i]);
}

// out = (addin ? addin : 0) + g[c]*((x+t)*BN_INV) + b[c]; optional f16 copy.
__global__ void bn_combine(const float* __restrict__ x, const float* __restrict__ t,
                           const float* __restrict__ addin, const float* __restrict__ g,
                           const float* __restrict__ b, float* __restrict__ out,
                           _Float16* __restrict__ out16, int n) {
    const int i = blockIdx.x * blockDim.x + threadIdx.x;
    if (i >= n) return;
    const int c = i & (DIM - 1);
    float v = (x[i] + t[i]) * BN_INV;
    v = g[c] * v + b[c];
    if (addin) v += addin[i];
    out[i] = v;
    if (out16) out16[i] = (_Float16)v;
}

// ---------------------------------------------------------------------------
static inline void run_gemm(const _Float16* A, const _Float16* W, const float* bias,
                            float* out32, _Float16* out16,
                            int Nrows, int K, int M, int relu, hipStream_t s) {
    dim3 grid(Nrows / 64, M / 64);
    gemm_wmma<<<grid, 128, 0, s>>>(A, W, bias, out32, out16, K, M, relu);
}

extern "C" void kernel_launch(void* const* d_in, const int* in_sizes, int n_in,
                              void* d_out, int out_size, void* d_ws, size_t ws_size,
                              hipStream_t stream) {
    const float* x   = (const float*)d_in[0];
    const float* ea  = (const float*)d_in[1];
    const int*   ei  = (const int*)  d_in[2];
    const float* gw1 = (const float*)d_in[3];
    const float* gb1 = (const float*)d_in[4];
    const float* gw2 = (const float*)d_in[5];
    const float* gb2 = (const float*)d_in[6];
    const float* ipw = (const float*)d_in[7];
    const float* ipb = (const float*)d_in[8];
    const float* opw = (const float*)d_in[9];
    const float* opb = (const float*)d_in[10];
    const float* g1l = (const float*)d_in[11];
    const float* b1l = (const float*)d_in[12];
    const float* g1g = (const float*)d_in[13];
    const float* b1g = (const float*)d_in[14];
    const float* g2  = (const float*)d_in[15];
    const float* b2  = (const float*)d_in[16];
    const float* fw1 = (const float*)d_in[17];
    const float* fb1 = (const float*)d_in[18];
    const float* fw2 = (const float*)d_in[19];
    const float* fb2 = (const float*)d_in[20];
    float* out = (float*)d_out;

    // Workspace (~164 MB total -> fits 192 MB L2):
    //   f32: F0 aggr | F1 t2/hg/f2 | F2 h_local | F3 h      (N*512 floats)
    //   f16: H0 z/qkv/f1 (N*384) | H1 t1/o/h16 (N*128) | H2 x16 (N*128) | HW weights
    const size_t N = NNODES;
    float* F0 = (float*)d_ws;
    float* F1 = F0 + N * 128;
    float* F2 = F1 + N * 128;
    float* F3 = F2 + N * 128;
    _Float16* H0 = (_Float16*)(F3 + N * 128);
    _Float16* H1 = H0 + N * 384;
    _Float16* H2 = H1 + N * 128;
    _Float16* HW = H2 + N * 128;
    _Float16* Wg1 = HW;                 // 128*128
    _Float16* Wg2 = Wg1 + 16384;        // 128*128
    _Float16* Wip = Wg2 + 16384;        // 384*128
    _Float16* Wop = Wip + 49152;        // 128*128
    _Float16* Wf1 = Wop + 16384;        // 256*128
    _Float16* Wf2 = Wf1 + 32768;        // 128*256

    const int TOT = (int)(N * DIM);     // 6,291,456 (multiple of 256)

    // ---- one-time f16 conversions (weights + x) ----
    cvt16<<<TOT / 256, 256, 0, stream>>>(x, H2, TOT);
    cvt16<<<64,  256, 0, stream>>>(gw1, Wg1, 16384);
    cvt16<<<64,  256, 0, stream>>>(gw2, Wg2, 16384);
    cvt16<<<192, 256, 0, stream>>>(ipw, Wip, 49152);
    cvt16<<<64,  256, 0, stream>>>(opw, Wop, 16384);
    cvt16<<<128, 256, 0, stream>>>(fw1, Wf1, 32768);
    cvt16<<<128, 256, 0, stream>>>(fw2, Wf2, 32768);

    // ---- GINE aggregation: z16 = f16(x + sum_j relu(x[src]+e)) ----
    zero_kernel <<<TOT / 256, 256, 0, stream>>>(F0, TOT);
    edge_scatter<<<ECNT / 8, 256, 0, stream>>>(x, ea, ei, F0);
    add2_f16    <<<TOT / 256, 256, 0, stream>>>(x, F0, H0, TOT);

    // ---- GIN MLP + BN residual -> h_local (F2) ----
    run_gemm(H0, Wg1, gb1, nullptr, H1, (int)N, 128, 128, 1, stream);   // t1 (f16)
    run_gemm(H1, Wg2, gb2, F1, nullptr, (int)N, 128, 128, 0, stream);   // t2 (f32)
    bn_combine<<<TOT / 256, 256, 0, stream>>>(x, F1, nullptr, g1l, b1l, F2, nullptr, TOT);

    // ---- attention branch ----
    run_gemm(H2, Wip, ipb, nullptr, H0, (int)N, 128, 384, 0, stream);   // qkv (f16)
    attn_kernel<<<GCNT * HEADS, NPG_, 0, stream>>>(H0, H1);             // o (f16)
    run_gemm(H1, Wop, opb, F1, nullptr, (int)N, 128, 128, 0, stream);   // hg (f32)

    // ---- h = h_local + bn1g(x + hg) -> F3 (f32) + H1 (f16) ----
    bn_combine<<<TOT / 256, 256, 0, stream>>>(x, F1, F2, g1g, b1g, F3, H1, TOT);

    // ---- FFN + final BN -> out ----
    run_gemm(H1, Wf1, fb1, nullptr, H0, (int)N, 128, 256, 1, stream);   // f1 (f16)
    run_gemm(H0, Wf2, fb2, F1, nullptr, (int)N, 256, 128, 0, stream);   // f2 (f32)
    bn_combine<<<TOT / 256, 256, 0, stream>>>(F3, F1, nullptr, g2, b2, out, nullptr, TOT);
}
